// SkelNet_89713276879117
// MI455X (gfx1250) — compile-verified, matrix operand order checked by reading, not run on hardware
//
#include <hip/hip_runtime.h>
#include <math.h>

// ---------------------------------------------------------------------------
// Tree-LSTM (SkelNet) for gfx1250: bf16 WMMA GEMM + fused gate epilogue.
//   PARENTS = [0,0,1,2,3,1,5,6,1,8,9,10,8,12,13,14]; NUM_LAYERS=2, H=1024,
//   B=256, D=512.  Per cell: g = [x|h_p] @ [Wih|Whh]^T + b  -> gates -> h,c.
// ---------------------------------------------------------------------------

typedef __attribute__((ext_vector_type(16))) __bf16 v16bf;
typedef __attribute__((ext_vector_type(8)))  __bf16 v8bf;
typedef __attribute__((ext_vector_type(8)))  float  v8f;

#define HDIM   1024
#define BDIM   256
#define NNODES 15
#define GATES4 4096   // 4*HDIM

// float -> bf16 bits, round-to-nearest-even
__device__ __forceinline__ unsigned short f2bf_bits(float f) {
    union { float f; unsigned u; } v; v.f = f;
    unsigned r = v.u + 0x7FFFu + ((v.u >> 16) & 1u);
    return (unsigned short)(r >> 16);
}

__device__ __forceinline__ float sigmoid_f(float x) {
    return 1.0f / (1.0f + __expf(-x));
}

// Load one 16x32 bf16 WMMA fragment (A-matrix layout; also valid for B when
// the weight matrix is stored row-major N x K, since column n of B == row n
// of W).  Per ISA 7.12.2: lanes 0-15 hold rows, K = {0..7,16..23}; lanes
// 16-31 same rows, K = {8..15,24..31}.  Two 16-byte loads per lane.
__device__ __forceinline__ v16bf load_frag16x32(const unsigned short* __restrict__ base,
                                                int ld, int row0, int k0, int lane) {
    int r  = lane & 15;
    int hi = lane >> 4;
    const __bf16* p = (const __bf16*)(base + (size_t)(row0 + r) * ld + k0 + hi * 8);
    v8bf lo = *reinterpret_cast<const v8bf*>(p);
    v8bf h8 = *reinterpret_cast<const v8bf*>(p + 16);
    return __builtin_shufflevector(lo, h8, 0,1,2,3,4,5,6,7,8,9,10,11,12,13,14,15);
}

// ------------------------- prep kernels ------------------------------------

__global__ void skel_convert_weights(const float* __restrict__ Wih,
                                     const float* __restrict__ Whh,
                                     const float* __restrict__ bih,
                                     const float* __restrict__ bhh,
                                     unsigned short* __restrict__ wih_bf,
                                     unsigned short* __restrict__ whh_bf,
                                     float* __restrict__ bias) {
    const size_t n = (size_t)2 * GATES4 * HDIM;
    for (size_t i = (size_t)blockIdx.x * blockDim.x + threadIdx.x; i < n;
         i += (size_t)gridDim.x * blockDim.x) {
        wih_bf[i] = f2bf_bits(Wih[i]);
        whh_bf[i] = f2bf_bits(Whh[i]);
    }
    const size_t nb = (size_t)2 * GATES4;
    for (size_t i = (size_t)blockIdx.x * blockDim.x + threadIdx.x; i < nb;
         i += (size_t)gridDim.x * blockDim.x) {
        bias[i] = bih[i] + bhh[i];
    }
}

__global__ void skel_build_x(const float* __restrict__ input,   // 15 x B x 512
                             const float* __restrict__ bridge,  // 15 x B x 512
                             unsigned short* __restrict__ xin)  // 15 x B x 1024 bf16
{
    const size_t n = (size_t)NNODES * BDIM * HDIM;
    for (size_t i = (size_t)blockIdx.x * blockDim.x + threadIdx.x; i < n;
         i += (size_t)gridDim.x * blockDim.x) {
        size_t jb = i / HDIM;          // node*B + b
        int    k  = (int)(i % HDIM);
        float  v  = (k < 512) ? input[jb * 512 + k] : bridge[jb * 512 + (k - 512)];
        xin[i] = f2bf_bits(v);
    }
}

__global__ void skel_zero_state(unsigned short* __restrict__ h0_bf,  // 2 x B x H
                                float* __restrict__ c0)              // 2 x B x H
{
    const size_t n = (size_t)2 * BDIM * HDIM;
    for (size_t i = (size_t)blockIdx.x * blockDim.x + threadIdx.x; i < n;
         i += (size_t)gridDim.x * blockDim.x) {
        h0_bf[i] = 0;
        c0[i]    = 0.0f;
    }
}

// ------------------------- fused LSTM cell ---------------------------------
// grid = (HDIM/64, BDIM/64); block = 256 threads = 8 waves (2 along M x 4
// along N).  Each wave: 32 batch rows x 16 hidden cols, all 4 gates.
__global__ __launch_bounds__(256)
void skel_lstm_cell_wmma(const unsigned short* __restrict__ x,      // B x H bf16
                         const unsigned short* __restrict__ hprev,  // B x H bf16
                         const unsigned short* __restrict__ Wih,    // 4H x H bf16
                         const unsigned short* __restrict__ Whh,    // 4H x H bf16
                         const float* __restrict__ bias,            // 4H
                         const float* __restrict__ cprev,           // B x H
                         unsigned short* __restrict__ hout_bf,      // B x H bf16
                         float* __restrict__ cout,                  // B x H
                         float* __restrict__ hout_f32)              // B x H or null
{
    const int lane  = threadIdx.x & 31;
    const int wave  = threadIdx.x >> 5;
    const int waveM = wave >> 2;                    // 0..1
    const int waveN = wave & 3;                     // 0..3
    const int m0    = blockIdx.y * 64 + waveM * 32; // batch-row base
    const int n0    = blockIdx.x * 64 + waveN * 16; // hidden-col base

    v8f acc[4][2];
#pragma unroll
    for (int g = 0; g < 4; ++g)
#pragma unroll
        for (int mi = 0; mi < 2; ++mi)
            acc[g][mi] = (v8f)(0.0f);

    // ---- x @ Wih^T : K = 0..1023 ----
    for (int k = 0; k < HDIM; k += 32) {
        v16bf a0 = load_frag16x32(x, HDIM, m0,      k, lane);
        v16bf a1 = load_frag16x32(x, HDIM, m0 + 16, k, lane);
#pragma unroll
        for (int g = 0; g < 4; ++g) {
            v16bf bm = load_frag16x32(Wih, HDIM, g * HDIM + n0, k, lane);
            acc[g][0] = __builtin_amdgcn_wmma_f32_16x16x32_bf16(
                false, a0, false, bm, (short)0, acc[g][0], false, false);
            acc[g][1] = __builtin_amdgcn_wmma_f32_16x16x32_bf16(
                false, a1, false, bm, (short)0, acc[g][1], false, false);
        }
    }
    // ---- h_prev @ Whh^T : K = 0..1023 ----
    for (int k = 0; k < HDIM; k += 32) {
        v16bf a0 = load_frag16x32(hprev, HDIM, m0,      k, lane);
        v16bf a1 = load_frag16x32(hprev, HDIM, m0 + 16, k, lane);
#pragma unroll
        for (int g = 0; g < 4; ++g) {
            v16bf bm = load_frag16x32(Whh, HDIM, g * HDIM + n0, k, lane);
            acc[g][0] = __builtin_amdgcn_wmma_f32_16x16x32_bf16(
                false, a0, false, bm, (short)0, acc[g][0], false, false);
            acc[g][1] = __builtin_amdgcn_wmma_f32_16x16x32_bf16(
                false, a1, false, bm, (short)0, acc[g][1], false, false);
        }
    }

    // ---- fused gate epilogue (fp32) ----
    // C/D layout: VGPR r, lanes 0-15 -> row m0+r,   col n0+lane
    //                     lanes 16-31-> row m0+8+r, col n0+lane-16
    const int ncol  = n0 + (lane & 15);
    const int rbase = (lane >> 4) << 3;
    const float bI = bias[          ncol];
    const float bF = bias[    HDIM + ncol];
    const float bG = bias[2 * HDIM + ncol];
    const float bO = bias[3 * HDIM + ncol];

#pragma unroll
    for (int mi = 0; mi < 2; ++mi) {
#pragma unroll
        for (int r = 0; r < 8; ++r) {
            const int    row = m0 + mi * 16 + rbase + r;
            const size_t idx = (size_t)row * HDIM + ncol;
            const float  cp  = cprev[idx];
            const float  iv  = sigmoid_f(acc[0][mi][r] + bI);
            const float  fv  = sigmoid_f(acc[1][mi][r] + bF);
            const float  gv  = tanhf(acc[2][mi][r] + bG);
            const float  ov  = sigmoid_f(acc[3][mi][r] + bO);
            const float  cn  = fv * cp + iv * gv;
            const float  hn  = ov * tanhf(cn);
            cout[idx]    = cn;
            hout_bf[idx] = f2bf_bits(hn);
            if (hout_f32) hout_f32[idx] = hn;
        }
    }
}

// ------------------------- host orchestration ------------------------------

extern "C" void kernel_launch(void* const* d_in, const int* in_sizes, int n_in,
                              void* d_out, int out_size, void* d_ws, size_t ws_size,
                              hipStream_t stream) {
    (void)in_sizes; (void)n_in; (void)out_size; (void)ws_size;
    static const int PAR[16] = {0, 0, 1, 2, 3, 1, 5, 6, 1, 8, 9, 10, 8, 12, 13, 14};

    const float* input  = (const float*)d_in[0]; // 15 x 256 x 512
    const float* bridge = (const float*)d_in[1]; // 15 x 256 x 512
    const float* W_ih   = (const float*)d_in[2]; // 2 x 4096 x 1024
    const float* W_hh   = (const float*)d_in[3]; // 2 x 4096 x 1024
    const float* b_ih   = (const float*)d_in[4]; // 2 x 4096
    const float* b_hh   = (const float*)d_in[5]; // 2 x 4096
    float*       out    = (float*)d_out;         // 15 x 256 x 1024

    // workspace carve-up (256B aligned)
    char*  ws  = (char*)d_ws;
    size_t off = 0;
    auto carve = [&](size_t bytes) -> void* {
        off = (off + 255) & ~(size_t)255;
        void* p = ws + off;
        off += bytes;
        return p;
    };
    const size_t wElems = (size_t)2 * GATES4 * HDIM;           // per weight tensor
    const size_t sElems = (size_t)BDIM * HDIM;                 // one (B,H) slab
    unsigned short* wih_bf = (unsigned short*)carve(wElems * 2);
    unsigned short* whh_bf = (unsigned short*)carve(wElems * 2);
    float*          bias   = (float*)carve((size_t)2 * GATES4 * 4);
    unsigned short* xin    = (unsigned short*)carve((size_t)NNODES * sElems * 2);
    unsigned short* hbf    = (unsigned short*)carve((size_t)16 * 2 * sElems * 2); // [node][layer][B][H]
    float*          cst    = (float*)carve((size_t)16 * 2 * sElems * 4);          // [node][layer][B][H]

    auto hPtr = [&](int node, int layer) {
        return hbf + ((size_t)node * 2 + layer) * sElems;
    };
    auto cPtr = [&](int node, int layer) {
        return cst + ((size_t)node * 2 + layer) * sElems;
    };

    // one-time prep (re-run every call: deterministic, graph-capture safe)
    skel_convert_weights<<<2048, 256, 0, stream>>>(W_ih, W_hh, b_ih, b_hh,
                                                   wih_bf, whh_bf, bias);
    skel_build_x<<<2048, 256, 0, stream>>>(input, bridge, xin);
    skel_zero_state<<<1024, 256, 0, stream>>>(hPtr(0, 0), cPtr(0, 0));

    const dim3 grid(HDIM / 64, BDIM / 64);  // 16 x 4
    const dim3 block(256);
    for (int j = 1; j <= NNODES; ++j) {
        const int p = PAR[j];
        for (int l = 0; l < 2; ++l) {
            const unsigned short* xl =
                (l == 0) ? (xin + (size_t)(j - 1) * sElems) : hPtr(j, 0);
            float* of = (l == 1) ? (out + (size_t)(j - 1) * sElems) : nullptr;
            skel_lstm_cell_wmma<<<grid, block, 0, stream>>>(
                xl, hPtr(p, l),
                wih_bf + (size_t)l * GATES4 * HDIM,
                whh_bf + (size_t)l * GATES4 * HDIM,
                bias + (size_t)l * GATES4,
                cPtr(p, l), hPtr(j, l), cPtr(j, l), of);
        }
    }
}
